// DiTBlock_42090679501019
// MI455X (gfx1250) — 8-bit matrix operand order repaired and checked by reading, compiled, not run on hardware
//
#include <hip/hip_runtime.h>
#include <hip/hip_bf16.h>

typedef __attribute__((ext_vector_type(8))) int v8i;

#define BN 256

__device__ __forceinline__ float sigmoidf_(float x) { return 1.0f / (1.0f + __expf(-x)); }
__device__ __forceinline__ float siluf_(float x)    { return x * sigmoidf_(x); }

__device__ __forceinline__ float blockReduceSum(float v, float* red) {
    int tid = threadIdx.x;
    red[tid] = v; __syncthreads();
    for (int s = BN / 2; s > 0; s >>= 1) {
        if (tid < s) red[tid] += red[tid + s];
        __syncthreads();
    }
    float r = red[0]; __syncthreads();
    return r;
}
__device__ __forceinline__ float blockReduceMax(float v, float* red) {
    int tid = threadIdx.x;
    red[tid] = v; __syncthreads();
    for (int s = BN / 2; s > 0; s >>= 1) {
        if (tid < s) red[tid] = fmaxf(red[tid], red[tid + s]);
        __syncthreads();
    }
    float r = red[0]; __syncthreads();
    return r;
}

__device__ __forceinline__ void async_tile32(unsigned lds, unsigned voff, unsigned long long base) {
    asm volatile(
        "global_load_async_to_lds_b128 %0, %1, %2\n\t"
        "global_load_async_to_lds_b128 %0, %1, %2 offset:16"
        :: "v"(lds), "v"(voff), "s"(base) : "memory");
}

__global__ __launch_bounds__(BN) void k_abs_partial(const float* __restrict__ W, long n,
                                                    float* __restrict__ partial) {
    __shared__ float red[BN];
    float s = 0.f;
    for (long i = (long)blockIdx.x * BN + threadIdx.x; i < n; i += (long)gridDim.x * BN)
        s += fabsf(W[i]);
    s = blockReduceSum(s, red);
    if (threadIdx.x == 0) partial[blockIdx.x] = s;
}

__global__ __launch_bounds__(BN) void k_abs_final(const float* __restrict__ partial, int nb,
                                                  long n, float* __restrict__ slot) {
    __shared__ float red[BN];
    float s = 0.f;
    for (int i = threadIdx.x; i < nb; i += BN) s += partial[i];
    s = blockReduceSum(s, red);
    if (threadIdx.x == 0) {
        float mean = s / (float)n;
        slot[0] = fmaxf(mean, 1e-5f);
    }
}

__global__ __launch_bounds__(BN) void k_wquant(const float* __restrict__ W,
                                               const float* __restrict__ slot,
                                               signed char* __restrict__ Q, long n) {
    float inv = 1.0f / slot[0];
    for (long i = (long)blockIdx.x * BN + threadIdx.x; i < n; i += (long)gridDim.x * BN) {
        float q = rintf(W[i] * inv);
        q = fmaxf(-1.f, fminf(1.f, q));
        Q[i] = (signed char)(int)q;
    }
}

__global__ __launch_bounds__(BN) void k_rms_quant(const float* __restrict__ X,
                                                  const float* __restrict__ nw,
                                                  signed char* __restrict__ Q,
                                                  float* __restrict__ A,
                                                  int K, float eps) {
    __shared__ float red[BN];
    int t = blockIdx.x;
    const float* xr = X + (size_t)t * K;
    float ss = 0.f;
    for (int k = threadIdx.x; k < K; k += BN) { float v = xr[k]; ss += v * v; }
    ss = blockReduceSum(ss, red);
    float r = rsqrtf(ss / (float)K + eps);
    float am = 0.f;
    for (int k = threadIdx.x; k < K; k += BN) am = fmaxf(am, fabsf(xr[k] * r * nw[k]));
    am = blockReduceMax(am, red);
    float amax = fmaxf(am, 1e-5f);
    float scl = 127.f / amax;
    for (int k = threadIdx.x; k < K; k += BN) {
        int q = (int)rintf(xr[k] * r * nw[k] * scl);
        q = max(-128, min(127, q));
        Q[(size_t)t * K + k] = (signed char)q;
    }
    if (threadIdx.x == 0) A[t] = amax / 127.f;
}

__global__ __launch_bounds__(BN) void k_rmsnorm_w(const float* __restrict__ X,
                                                  const float* __restrict__ w,
                                                  float* __restrict__ Y, int K, float eps) {
    __shared__ float red[BN];
    int t = blockIdx.x;
    const float* xr = X + (size_t)t * K;
    float ss = 0.f;
    for (int k = threadIdx.x; k < K; k += BN) { float v = xr[k]; ss += v * v; }
    ss = blockReduceSum(ss, red);
    float r = rsqrtf(ss / (float)K + eps);
    for (int k = threadIdx.x; k < K; k += BN)
        Y[(size_t)t * K + k] = xr[k] * r * w[k];
}

__global__ __launch_bounds__(BN) void k_modln_quant3(
    const float* __restrict__ X, const float* __restrict__ Sc, const float* __restrict__ Sh,
    int prow,
    const float* n0p, const float* n1p, const float* n2p,
    signed char* q0, signed char* q1, signed char* q2,
    float* a0, float* a1, float* a2,
    int D, int Lseq, float lneps) {
    __shared__ float mxs[1024];
    __shared__ float red[BN];
    int t = blockIdx.x;
    int b = t / Lseq;
    const float* xr = X + (size_t)t * D;

    float s = 0.f;
    for (int k = threadIdx.x; k < D; k += BN) s += xr[k];
    s = blockReduceSum(s, red);
    float mean = s / (float)D;
    float v = 0.f;
    for (int k = threadIdx.x; k < D; k += BN) { float d = xr[k] - mean; v += d * d; }
    v = blockReduceSum(v, red);
    float rv = rsqrtf(v / (float)D + lneps);
    for (int k = threadIdx.x; k < D; k += BN)
        mxs[k] = (xr[k] - mean) * rv * (1.f + Sc[(size_t)b * prow + k]) + Sh[(size_t)b * prow + k];
    __syncthreads();

    float ss = 0.f;
    for (int k = threadIdx.x; k < D; k += BN) ss += mxs[k] * mxs[k];
    ss = blockReduceSum(ss, red);
    float r8 = rsqrtf(ss / (float)D + 1e-8f);

    const float* nps[3] = { n0p, n1p, n2p };
    signed char* qs[3]  = { q0, q1, q2 };
    float* as_[3]       = { a0, a1, a2 };
    for (int hh = 0; hh < 3; ++hh) {
        const float* np = nps[hh];
        if (!np) continue;
        float am = 0.f;
        for (int k = threadIdx.x; k < D; k += BN) am = fmaxf(am, fabsf(mxs[k] * r8 * np[k]));
        am = blockReduceMax(am, red);
        float amax = fmaxf(am, 1e-5f);
        float scl = 127.f / amax;
        for (int k = threadIdx.x; k < D; k += BN) {
            int q = (int)rintf(mxs[k] * r8 * np[k] * scl);
            q = max(-128, min(127, q));
            qs[hh][(size_t)t * D + k] = (signed char)q;
        }
        if (threadIdx.x == 0) as_[hh][t] = amax / 127.f;
        __syncthreads();
    }
}

__global__ __launch_bounds__(256) void k_gemm_i8_wmma(
    const signed char* __restrict__ XQ, const float* __restrict__ Ascl,
    const signed char* __restrict__ WQ, const float* __restrict__ swp,
    const float* __restrict__ bias, const float* __restrict__ resid,
    const float* __restrict__ gmod, int grow,
    float* __restrict__ Y, int T, int O, int K, int Lseq, int out_bf16) {
    __shared__ __align__(16) signed char lA[2][128 * 64];
    __shared__ __align__(16) signed char lB[2][128 * 64];

    const int tid = threadIdx.x;
    const int lane = tid & 31;
    const int w = tid >> 5;
    const int wm = w & 3;
    const int wn = w >> 2;
    const int m0 = blockIdx.x * 128;
    const int n0 = blockIdx.y * 128;

    v8i acc[8];
#pragma unroll
    for (int i = 0; i < 8; ++i) acc[i] = (v8i){0, 0, 0, 0, 0, 0, 0, 0};

    const int srow = tid >> 1;
    const int scol = (tid & 1) * 32;
    int garow = m0 + srow;
    if (garow >= T) garow = 0;
    const unsigned aoff0 = (unsigned)((size_t)garow * K + scol);
    const unsigned boff0 = (unsigned)((size_t)(n0 + srow) * K + scol);
    const unsigned ldsAb[2] = { (unsigned)(uintptr_t)&lA[0][srow * 64 + scol],
                                (unsigned)(uintptr_t)&lA[1][srow * 64 + scol] };
    const unsigned ldsBb[2] = { (unsigned)(uintptr_t)&lB[0][srow * 64 + scol],
                                (unsigned)(uintptr_t)&lB[1][srow * 64 + scol] };
    const unsigned long long abase = (unsigned long long)XQ;
    const unsigned long long bbase = (unsigned long long)WQ;

    async_tile32(ldsAb[0], aoff0, abase);
    async_tile32(ldsBb[0], boff0, bbase);

    int cur = 0;
    for (int k0 = 0; k0 < K; k0 += 64) {
        if (k0 + 64 < K) {
            const unsigned av = aoff0 + (unsigned)(k0 + 64);
            const unsigned bv = boff0 + (unsigned)(k0 + 64);
            async_tile32(ldsAb[cur ^ 1], av, abase);
            async_tile32(ldsBb[cur ^ 1], bv, bbase);
            if (k0 + 128 < K) {
                __builtin_prefetch((const void*)(XQ + (size_t)aoff0 + k0 + 128), 0, 1);
                __builtin_prefetch((const void*)(WQ + (size_t)boff0 + k0 + 128), 0, 1);
            }
            asm volatile("s_wait_asynccnt 0x4" ::: "memory");
        } else {
            asm volatile("s_wait_asynccnt 0x0" ::: "memory");
        }
        __syncthreads();

        const signed char* lAc = &lA[cur][0];
        const signed char* lBc = &lB[cur][0];
        const int kb  = (lane >> 4) * 8;
        const int kb2 = (lane >> 4) * 16;
        v8i afrag[2];
#pragma unroll
        for (int sub = 0; sub < 2; ++sub) {
            const int ar = wm * 32 + sub * 16 + (lane & 15);
            const signed char* ap = lAc + ar * 64 + kb2;
            int2 a01 = *(const int2*)(ap);
            int2 a23 = *(const int2*)(ap + 8);
            int2 a45 = *(const int2*)(ap + 32);
            int2 a67 = *(const int2*)(ap + 40);
            afrag[sub][0] = a01.x; afrag[sub][1] = a01.y;
            afrag[sub][2] = a23.x; afrag[sub][3] = a23.y;
            afrag[sub][4] = a45.x; afrag[sub][5] = a45.y;
            afrag[sub][6] = a67.x; afrag[sub][7] = a67.y;
        }
#pragma unroll
        for (int ns = 0; ns < 4; ++ns) {
            const int col = wn * 64 + ns * 16 + (lane & 15);
            const signed char* bp = lBc + col * 64 + kb2;
            int4 b03 = *(const int4*)(bp);
            int4 b47 = *(const int4*)(bp + 32);
            v8i bfrag;
            bfrag[0] = b03.x; bfrag[1] = b03.y; bfrag[2] = b03.z; bfrag[3] = b03.w;
            bfrag[4] = b47.x; bfrag[5] = b47.y; bfrag[6] = b47.z; bfrag[7] = b47.w;
#pragma unroll
            for (int sub = 0; sub < 2; ++sub) {
                acc[sub * 4 + ns] = __builtin_amdgcn_wmma_i32_16x16x64_iu8(
                    true, afrag[sub], true, bfrag, acc[sub * 4 + ns], false, false);
            }
        }
        __syncthreads();
        cur ^= 1;
    }

    const float sw = swp[0];
    const int rloc = (lane >> 4) * 8;
    const int colbase = n0 + wn * 64 + (lane & 15);
#pragma unroll
    for (int sub = 0; sub < 2; ++sub) {
#pragma unroll
        for (int ns = 0; ns < 4; ++ns) {
            const int col = colbase + ns * 16;
#pragma unroll
            for (int r = 0; r < 8; ++r) {
                const int row = m0 + wm * 32 + sub * 16 + r + rloc;
                if (row < T) {
                    float v = (float)acc[sub * 4 + ns][r] * Ascl[row] * sw;
                    if (bias) v += bias[col];
                    if (gmod) v *= gmod[(size_t)(row / Lseq) * grow + col];
                    const size_t idx = (size_t)row * O + col;
                    if (resid) v += resid[idx];
                    if (out_bf16) ((__hip_bfloat16*)Y)[idx] = __float2bfloat16(v);
                    else Y[idx] = v;
                }
            }
        }
    }
}

__global__ __launch_bounds__(BN) void k_silu(const float* __restrict__ X, float* __restrict__ Y, long n) {
    long i = (long)blockIdx.x * BN + threadIdx.x;
    if (i < n) Y[i] = siluf_(X[i]);
}

__global__ __launch_bounds__(BN) void k_combine_if(float* __restrict__ yi, float* __restrict__ yf, long n) {
    long i = (long)blockIdx.x * BN + threadIdx.x;
    if (i < n) {
        float f = sigmoidf_(yf[i]);
        yi[i] = siluf_(yi[i]) * (1.f - f);
        yf[i] = f;
    }
}

__global__ __launch_bounds__(BN) void k_swiglu(const float* __restrict__ X, float* __restrict__ Y,
                                               int R, int H) {
    long i = (long)blockIdx.x * BN + threadIdx.x;
    long n = (long)R * H;
    if (i < n) {
        int r = (int)(i / H);
        int k = (int)(i % H);
        float g = X[(size_t)r * 2 * H + k];
        float v = X[(size_t)r * 2 * H + H + k];
        Y[i] = siluf_(g) * v;
    }
}

__global__ __launch_bounds__(BN) void k_scan_chunk(const float* __restrict__ iv, const float* __restrict__ fv,
                                                   float* __restrict__ Fc, float* __restrict__ Hc,
                                                   int Bb, int Lx, int Dx, int CH) {
    const int nchunk = Lx / CH;
    long gid = (long)blockIdx.x * BN + threadIdx.x;
    long total = (long)Bb * nchunk * Dx;
    if (gid >= total) return;
    int d = (int)(gid % Dx);
    int c = (int)((gid / Dx) % nchunk);
    int b = (int)(gid / ((long)Dx * nchunk));
    float F = 1.f, H = 0.f;
    for (int s = 0; s < CH; ++s) {
        size_t idx = ((size_t)b * Lx + (size_t)c * CH + s) * Dx + d;
        float f = fv[idx];
        H = f * H + iv[idx];
        F *= f;
    }
    Fc[gid] = F; Hc[gid] = H;
}

__global__ __launch_bounds__(BN) void k_scan_carry(const float* __restrict__ Fc, float* __restrict__ Hc,
                                                   int Bb, int nchunk, int Dx) {
    long gid = (long)blockIdx.x * BN + threadIdx.x;
    long total = (long)Bb * Dx;
    if (gid >= total) return;
    int d = (int)(gid % Dx);
    int b = (int)(gid / Dx);
    float P = 0.f;
    for (int c = 0; c < nchunk; ++c) {
        size_t idx = ((size_t)b * nchunk + c) * Dx + d;
        float f = Fc[idx], hh = Hc[idx];
        Hc[idx] = P;
        P = f * P + hh;
    }
}

__global__ __launch_bounds__(BN) void k_scan_apply(const float* __restrict__ iv, const float* __restrict__ fv,
                                                   const float* __restrict__ Hc, float* __restrict__ hout,
                                                   int Bb, int Lx, int Dx, int CH) {
    const int nchunk = Lx / CH;
    long gid = (long)blockIdx.x * BN + threadIdx.x;
    long total = (long)Bb * nchunk * Dx;
    if (gid >= total) return;
    int d = (int)(gid % Dx);
    int c = (int)((gid / Dx) % nchunk);
    int b = (int)(gid / ((long)Dx * nchunk));
    float P = Hc[gid];
    for (int s = 0; s < CH; ++s) {
        size_t idx = ((size_t)b * Lx + (size_t)c * CH + s) * Dx + d;
        P = fv[idx] * P + iv[idx];
        hout[idx] = P;
    }
}

__global__ __launch_bounds__(BN) void k_ogate_quant(const float* __restrict__ h, const float* __restrict__ gn,
                                                    const float* __restrict__ yg, const float* __restrict__ now,
                                                    signed char* __restrict__ Q, float* __restrict__ A, int D) {
    __shared__ float os[1024];
    __shared__ float red[BN];
    int t = blockIdx.x;
    const float* hr = h + (size_t)t * D;
    const float* gr_ = yg + (size_t)t * D;
    float ss = 0.f;
    for (int k = threadIdx.x; k < D; k += BN) { float v = hr[k]; ss += v * v; }
    ss = blockReduceSum(ss, red);
    float r5 = rsqrtf(ss / (float)D + 1e-5f);
    for (int k = threadIdx.x; k < D; k += BN)
        os[k] = hr[k] * r5 * gn[k] * siluf_(gr_[k]);
    __syncthreads();
    float s2 = 0.f;
    for (int k = threadIdx.x; k < D; k += BN) s2 += os[k] * os[k];
    s2 = blockReduceSum(s2, red);
    float r8 = rsqrtf(s2 / (float)D + 1e-8f);
    float am = 0.f;
    for (int k = threadIdx.x; k < D; k += BN) am = fmaxf(am, fabsf(os[k] * r8 * now[k]));
    am = blockReduceMax(am, red);
    float amax = fmaxf(am, 1e-5f);
    float scl = 127.f / amax;
    for (int k = threadIdx.x; k < D; k += BN) {
        int q = (int)rintf(os[k] * r8 * now[k] * scl);
        q = max(-128, min(127, q));
        Q[(size_t)t * D + k] = (signed char)q;
    }
    if (threadIdx.x == 0) A[t] = amax / 127.f;
}

__global__ __launch_bounds__(BN) void k_swiglu_quant(const __hip_bfloat16* __restrict__ hid,
                                                     const float* __restrict__ nm2,
                                                     signed char* __restrict__ Q, float* __restrict__ A,
                                                     int H) {
    __shared__ float ssh[4096];
    __shared__ float red[BN];
    int t = blockIdx.x;
    const __hip_bfloat16* hr = hid + (size_t)t * 2 * H;
    for (int k = threadIdx.x; k < H; k += BN) {
        float g = __bfloat162float(hr[k]);
        float v = __bfloat162float(hr[H + k]);
        ssh[k] = siluf_(g) * v;
    }
    __syncthreads();
    float ss = 0.f;
    for (int k = threadIdx.x; k < H; k += BN) ss += ssh[k] * ssh[k];
    ss = blockReduceSum(ss, red);
    float r8 = rsqrtf(ss / (float)H + 1e-8f);
    float am = 0.f;
    for (int k = threadIdx.x; k < H; k += BN) am = fmaxf(am, fabsf(ssh[k] * r8 * nm2[k]));
    am = blockReduceMax(am, red);
    float amax = fmaxf(am, 1e-5f);
    float scl = 127.f / amax;
    for (int k = threadIdx.x; k < H; k += BN) {
        int q = (int)rintf(ssh[k] * r8 * nm2[k] * scl);
        q = max(-128, min(127, q));
        Q[(size_t)t * H + k] = (signed char)q;
    }
    if (threadIdx.x == 0) A[t] = amax / 127.f;
}

extern "C" void kernel_launch(void* const* d_in, const int* in_sizes, int n_in,
                              void* d_out, int out_size, void* d_ws, size_t ws_size,
                              hipStream_t stream) {
    (void)in_sizes; (void)n_in; (void)out_size; (void)ws_size;
    const int Bb = 4, Lx = 4096, Dx = 1024;
    const int T = Bb * Lx;
    const int MH = 4096, H2 = 8192;
    const int ADA = 2816, ADA2 = 5632;
    const int O6 = 6144;
    const int CH = 64, NC = Lx / CH;

    const float* x   = (const float*)d_in[0];
    const float* c   = (const float*)d_in[1];
    const float* ni  = (const float*)d_in[3];
    const float* nf  = (const float*)d_in[5];
    const float* ng  = (const float*)d_in[7];
    const float* now = (const float*)d_in[9];
    const float* gn  = (const float*)d_in[10];
    const float* nm1 = (const float*)d_in[12];
    const float* nm2 = (const float*)d_in[14];
    const float* na0 = (const float*)d_in[16];
    const float* na1 = (const float*)d_in[18];
    const float* na2 = (const float*)d_in[20];
    const float* ba3 = (const float*)d_in[22];
    const float* na3 = (const float*)d_in[23];
    const float* wr  = (const float*)d_in[24];
    const float* ba4 = (const float*)d_in[26];
    const float* na4 = (const float*)d_in[27];

    size_t off = 0;
    auto alloc = [&](size_t n) { off = (off + 255) & ~(size_t)255; size_t o = off; off += n; return o; };
    char* ws = (char*)d_ws;

    struct WSpec { int idx; int O; int K; };
    const WSpec wspec[11] = {
        {2, 1024, 1024}, {4, 1024, 1024}, {6, 1024, 1024}, {8, 1024, 1024},
        {11, H2, 1024}, {13, 1024, MH},
        {15, 1024, 1024}, {17, ADA2, 1024}, {19, 1024, ADA}, {21, O6, 1024}, {25, O6, O6}
    };
    size_t wq_off[11];
    for (int i = 0; i < 11; ++i) wq_off[i] = alloc((size_t)wspec[i].O * wspec[i].K);
    size_t scales_off  = alloc(64 * 4);
    size_t partial_off = alloc(1024 * 4);

    size_t cc_off   = alloc((size_t)Bb * Dx * 4);
    size_t q8c_off  = alloc((size_t)Bb * O6);
    size_t ac_off   = alloc(Bb * 4);
    size_t h0_off   = alloc((size_t)Bb * Dx * 4);
    size_t y1c_off  = alloc((size_t)Bb * ADA2 * 4);
    size_t sg_off   = alloc((size_t)Bb * ADA * 4);
    size_t h1_off   = alloc((size_t)Bb * Dx * 4);
    size_t h2_off   = alloc((size_t)Bb * O6 * 4);
    size_t h3_off   = alloc((size_t)Bb * O6 * 4);
    size_t par_off  = alloc((size_t)Bb * O6 * 4);

    size_t xqi_off = alloc((size_t)T * Dx);
    size_t xqf_off = alloc((size_t)T * Dx);
    size_t xqg_off = alloc((size_t)T * Dx);
    size_t ai_off  = alloc((size_t)T * 4);
    size_t af_off  = alloc((size_t)T * 4);
    size_t ag_off  = alloc((size_t)T * 4);
    size_t yi_off  = alloc((size_t)T * Dx * 4);
    size_t yf_off  = alloc((size_t)T * Dx * 4);
    size_t yg_off  = alloc((size_t)T * Dx * 4);
    size_t h_off   = alloc((size_t)T * Dx * 4);
    size_t fc_off  = alloc((size_t)Bb * NC * Dx * 4);
    size_t hc_off  = alloc((size_t)Bb * NC * Dx * 4);
    size_t oq_off  = alloc((size_t)T * Dx);
    size_t ao_off  = alloc((size_t)T * 4);
    size_t xm1_off = alloc((size_t)T * Dx);
    size_t am1_off = alloc((size_t)T * 4);
    size_t xm2_off = alloc((size_t)T * MH);
    size_t am2_off = alloc((size_t)T * 4);
    (void)yf_off; (void)yg_off;

    float* scales = (float*)(ws + scales_off);
    float* partial = (float*)(ws + partial_off);
    float* params = (float*)(ws + par_off);
    float* fout = (float*)d_out;

    for (int i = 0; i < 11; ++i) {
        long n = (long)wspec[i].O * wspec[i].K;
        const float* W = (const float*)d_in[wspec[i].idx];
        k_abs_partial<<<1024, BN, 0, stream>>>(W, n, partial);
        k_abs_final<<<1, BN, 0, stream>>>(partial, 1024, n, scales + i);
        long nb = (n + BN - 1) / BN;
        k_wquant<<<(int)min(nb, (long)65535), BN, 0, stream>>>(W, scales + i,
                                                               (signed char*)(ws + wq_off[i]), n);
    }

    auto gemm = [&](size_t xq, size_t ascl, int wi_, const float* bias, const float* resid,
                    const float* gmod, int grow, void* Y, int Tt, int Oo, int Kk, int out_bf16) {
        dim3 g((Tt + 127) / 128, Oo / 128);
        k_gemm_i8_wmma<<<g, 256, 0, stream>>>(
            (const signed char*)(ws + xq), (const float*)(ws + ascl),
            (const signed char*)(ws + wq_off[wi_]), scales + wi_,
            bias, resid, gmod, grow, (float*)Y, Tt, Oo, Kk, (gmod ? Lx : 1), out_bf16);
    };

    k_silu<<<(Bb * Dx + BN - 1) / BN, BN, 0, stream>>>(c, (float*)(ws + cc_off), (long)Bb * Dx);
    k_rms_quant<<<Bb, BN, 0, stream>>>((const float*)(ws + cc_off), na0,
                                       (signed char*)(ws + q8c_off), (float*)(ws + ac_off), Dx, 1e-8f);
    gemm(q8c_off, ac_off, 6, nullptr, nullptr, nullptr, 0, ws + h0_off, Bb, Dx, Dx, 0);

    k_rms_quant<<<Bb, BN, 0, stream>>>((const float*)(ws + h0_off), na1,
                                       (signed char*)(ws + q8c_off), (float*)(ws + ac_off), Dx, 1e-8f);
    gemm(q8c_off, ac_off, 7, nullptr, nullptr, nullptr, 0, ws + y1c_off, Bb, ADA2, Dx, 0);

    k_swiglu<<<(Bb * ADA + BN - 1) / BN, BN, 0, stream>>>((const float*)(ws + y1c_off),
                                                          (float*)(ws + sg_off), Bb, ADA);
    k_rms_quant<<<Bb, BN, 0, stream>>>((const float*)(ws + sg_off), na2,
                                       (signed char*)(ws + q8c_off), (float*)(ws + ac_off), ADA, 1e-8f);
    gemm(q8c_off, ac_off, 8, nullptr, nullptr, nullptr, 0, ws + h1_off, Bb, Dx, ADA, 0);

    k_rms_quant<<<Bb, BN, 0, stream>>>((const float*)(ws + h1_off), na3,
                                       (signed char*)(ws + q8c_off), (float*)(ws + ac_off), Dx, 1e-8f);
    gemm(q8c_off, ac_off, 9, ba3, nullptr, nullptr, 0, ws + h2_off, Bb, O6, Dx, 0);

    k_rmsnorm_w<<<Bb, BN, 0, stream>>>((const float*)(ws + h2_off), wr, (float*)(ws + h3_off), O6, 1e-6f);
    k_rms_quant<<<Bb, BN, 0, stream>>>((const float*)(ws + h3_off), na4,
                                       (signed char*)(ws + q8c_off), (float*)(ws + ac_off), O6, 1e-8f);
    gemm(q8c_off, ac_off, 10, ba4, nullptr, nullptr, 0, params, Bb, O6, O6, 0);

    k_modln_quant3<<<T, BN, 0, stream>>>(
        x, params + 1024, params + 0, O6, ni, nf, ng,
        (signed char*)(ws + xqi_off), (signed char*)(ws + xqf_off), (signed char*)(ws + xqg_off),
        (float*)(ws + ai_off), (float*)(ws + af_off), (float*)(ws + ag_off),
        Dx, Lx, 1e-6f);

    gemm(xqi_off, ai_off, 0, nullptr, nullptr, nullptr, 0, ws + yi_off, T, Dx, Dx, 0);
    gemm(xqf_off, af_off, 1, nullptr, nullptr, nullptr, 0, ws + yf_off, T, Dx, Dx, 0);
    gemm(xqg_off, ag_off, 2, nullptr, nullptr, nullptr, 0, ws + yg_off, T, Dx, Dx, 0);

    long nelem = (long)T * Dx;
    k_combine_if<<<(int)((nelem + BN - 1) / BN), BN, 0, stream>>>((float*)(ws + yi_off),
                                                                 (float*)(ws + yf_off), nelem);
    long nsc = (long)Bb * NC * Dx;
    k_scan_chunk<<<(int)((nsc + BN - 1) / BN), BN, 0, stream>>>(
        (const float*)(ws + yi_off), (const float*)(ws + yf_off),
        (float*)(ws + fc_off), (float*)(ws + hc_off), Bb, Lx, Dx, CH);
    k_scan_carry<<<(Bb * Dx + BN - 1) / BN, BN, 0, stream>>>(
        (const float*)(ws + fc_off), (float*)(ws + hc_off), Bb, NC, Dx);
    k_scan_apply<<<(int)((nsc + BN - 1) / BN), BN, 0, stream>>>(
        (const float*)(ws + yi_off), (const float*)(ws + yf_off), (const float*)(ws + hc_off),
        (float*)(ws + h_off), Bb, Lx, Dx, CH);

    k_ogate_quant<<<T, BN, 0, stream>>>((const float*)(ws + h_off), gn, (const float*)(ws + yg_off),
                                        now, (signed char*)(ws + oq_off), (float*)(ws + ao_off), Dx);
    gemm(oq_off, ao_off, 3, nullptr, x, params + 2048, O6, fout, T, Dx, Dx, 0);

    k_modln_quant3<<<T, BN, 0, stream>>>(
        fout, params + 4096, params + 3072, O6, nm1, nullptr, nullptr,
        (signed char*)(ws + xm1_off), nullptr, nullptr,
        (float*)(ws + am1_off), nullptr, nullptr,
        Dx, Lx, 1e-6f);

    __hip_bfloat16* hidden = (__hip_bfloat16*)(ws + yi_off);
    gemm(xm1_off, am1_off, 4, nullptr, nullptr, nullptr, 0, hidden, T, H2, Dx, 1);

    k_swiglu_quant<<<T, BN, 0, stream>>>(hidden, nm2, (signed char*)(ws + xm2_off),
                                         (float*)(ws + am2_off), MH);
    gemm(xm2_off, am2_off, 5, nullptr, fout, params + 5120, O6, fout, T, Dx, MH, 0);
}
